// TrajModel_68934225101097
// MI455X (gfx1250) — matrix operand order repaired and checked by reading, not compiled, not run on hardware
//
#include <hip/hip_runtime.h>
#include <stdint.h>

// ---------------- CDNA5 WMMA types / helpers ----------------
typedef __attribute__((ext_vector_type(16))) __bf16 v16bf;
typedef __attribute__((ext_vector_type(8)))  float  v8f;

union Frag16 { v16bf v; uint4 q[2]; };
union Acc8   { v8f v; float f[8]; };

__device__ __forceinline__ unsigned short f2bf(float f) {
    unsigned u = __float_as_uint(f);
    u += 0x7fffu + ((u >> 16) & 1u);          // round-to-nearest-even
    return (unsigned short)(u >> 16);
}

__device__ __forceinline__ v8f wmma_bf16(v16bf a, v16bf b, v8f c) {
    // v_wmma_f32_16x16x32_bf16  D = A(16x32) * B(32x16) + C
    return __builtin_amdgcn_wmma_f32_16x16x32_bf16(false, a, false, b, (short)0, c, false, false);
}

// Model dims
constexpr int Bsz = 2048, Din = 64, Dm = 256, Gm = 768, Tn = 256;
constexpr int BT  = 32;          // batch rows per block
constexpr int STR = 264;         // LDS row stride (bf16 elems): 16B-aligned, bank-friendly

// A fragment (16x32 bf16) from LDS: lane&15 = row, lane>>4 selects K-chunk halves {k0..k0+7, k0+16..k0+23}.
__device__ __forceinline__ Frag16 loadA_lds(const unsigned short* sh, int row0, int k0, int c, int kh) {
    Frag16 f;
    const unsigned short* p = sh + (row0 + c) * STR + k0 + kh * 8;
    f.q[0] = *(const uint4*)p;
    f.q[1] = *(const uint4*)(p + 16);
    return f;
}
// B fragment (32x16 bf16) of W^T from row-major bf16 W[N][256]: lane&15 = output col (row of W),
// lane>>4 selects K-chunk halves {k0..k0+7, k0+16..k0+23} (+8 for upper lanes), matching loadA_lds.
__device__ __forceinline__ Frag16 loadB_w(const unsigned short* w, int n0, int k0, int c, int kh) {
    Frag16 f;
    const unsigned short* p = w + (size_t)(n0 + c) * Dm + k0 + kh * 8;
    f.q[0] = *(const uint4*)p;
    f.q[1] = *(const uint4*)(p + 16);
    return f;
}

// ---------------- prep kernels ----------------
__global__ void prep_weights(const float* __restrict__ w_hh, const float* __restrict__ w_ih_rest,
                             const float* __restrict__ lin_w,
                             unsigned short* __restrict__ whh, unsigned short* __restrict__ wih,
                             unsigned short* __restrict__ wlin) {
    int i = blockIdx.x * 256 + threadIdx.x;                 // 1,048,576 total
    if (i < 589824)            whh[i]           = f2bf(w_hh[i]);
    else if (i < 983040)       wih[i - 589824]  = f2bf(w_ih_rest[i - 589824]);
    else if (i < 1048576)      wlin[i - 983040] = f2bf(lin_w[i - 983040]);
}

__global__ void init_state(const float* __restrict__ h0, float* __restrict__ state) {
    int i = blockIdx.x * 256 + threadIdx.x;
    if (i < 3 * Bsz * Dm) state[i] = h0[i];
}

// gi0 = X @ w_ih0.T   (loop-invariant; bias added later uniformly)
__global__ void gi0_kernel(const float* __restrict__ X, const float* __restrict__ w_ih0,
                           float* __restrict__ gi0) {
    int i = blockIdx.x * 256 + threadIdx.x;
    if (i >= Bsz * Gm) return;
    int b = i / Gm, g = i % Gm;
    const float* xr = X + (size_t)b * Din;
    const float* wr = w_ih0 + (size_t)g * Din;
    float s = 0.f;
#pragma unroll 8
    for (int k = 0; k < Din; ++k) s += xr[k] * wr[k];
    gi0[i] = s;
}

// ---------------- main persistent GRU kernel ----------------
// (256, 4): request >=4 waves/SIMD so the compiler stays <=256 arch VGPRs
// (no s_set_vgpr_msb churn) and there is enough occupancy to hide the L2
// latency of the B-fragment global_load_b128 feeding each WMMA.
__global__ __launch_bounds__(256, 4)
void traj_main(const float* __restrict__ gi0,
               const float* __restrict__ b_ih, const float* __restrict__ b_hh,
               const float* __restrict__ lin_b,
               const float* __restrict__ hw1, const float* __restrict__ hb1,
               const float* __restrict__ hw2, const float* __restrict__ hb2,
               const unsigned short* __restrict__ whh,
               const unsigned short* __restrict__ wih,
               const unsigned short* __restrict__ wlin,
               float* __restrict__ state,
               float* __restrict__ out) {
    extern __shared__ __align__(16) unsigned char smem[];
    unsigned short* bufA = (unsigned short*)smem;          // BT x STR bf16 activations (double buffer)
    unsigned short* bufB = bufA + BT * STR;
    unsigned short* shH  = bufB + BT * STR;                // bf16 copy of current cell's h state

    const int tid   = threadIdx.x;
    const int lane  = tid & 31;
    const int wave  = tid >> 5;
    const int c     = lane & 15;
    const int kh    = lane >> 4;
    const int strip = wave & 1;        // which 16-row strip
    const int jpart = wave >> 1;       // which quarter of the 16 column-tiles
    const int mRow0 = strip * 16;
    const int rowBase = blockIdx.x * BT;

    for (int t = 0; t < Tn; ++t) {
        for (int cell = 0; cell < 3; ++cell) {
            const unsigned short* xin = (cell == 1) ? bufA : bufB;   // cell0 unused
            unsigned short* xout      = (cell == 1) ? bufB : bufA;
            float* st = state + (size_t)cell * Bsz * Dm;
            const unsigned short* Wh = whh + (size_t)cell * Gm * Dm;
            const unsigned short* Wi = (cell > 0) ? wih + (size_t)(cell - 1) * Gm * Dm : wih;

            __syncthreads();
            // stage bf16 h-state (each thread converts exactly the elements it owns/wrote)
            for (int v = 0; v < 8; ++v) {
                int r = mRow0 + v + 8 * kh;
                const float* srow = st + (size_t)(rowBase + r) * Dm;
                for (int j = 0; j < 4; ++j) {
                    int col = (jpart * 4 + j) * 16 + c;
                    shH[r * STR + col] = f2bf(srow[col]);
                }
            }
            __syncthreads();

#pragma unroll 1
            for (int j = 0; j < 4; ++j) {
                const int n0 = (jpart * 4 + j) * 16;       // column tile inside D
                Acc8 aR, aZ, aN, aH;                       // r, z, i_n, h_n accumulators
                if (cell == 0) {
                    for (int v = 0; v < 8; ++v) {
                        const float* g = gi0 + (size_t)(rowBase + mRow0 + v + 8 * kh) * Gm;
                        aR.f[v] = g[n0 + c];
                        aZ.f[v] = g[Dm + n0 + c];
                        aN.f[v] = g[2 * Dm + n0 + c];
                        aH.f[v] = 0.f;
                    }
                } else {
                    for (int v = 0; v < 8; ++v) { aR.f[v] = 0.f; aZ.f[v] = 0.f; aN.f[v] = 0.f; aH.f[v] = 0.f; }
                }
#pragma unroll 2
                for (int kk = 0; kk < 8; ++kk) {
                    const int k0 = kk * 32;
                    Frag16 ah = loadA_lds(shH, mRow0, k0, c, kh);
                    aR.v = wmma_bf16(ah.v, loadB_w(Wh, n0,          k0, c, kh).v, aR.v);
                    aZ.v = wmma_bf16(ah.v, loadB_w(Wh, Dm + n0,     k0, c, kh).v, aZ.v);
                    aH.v = wmma_bf16(ah.v, loadB_w(Wh, 2 * Dm + n0, k0, c, kh).v, aH.v);
                    if (cell > 0) {
                        Frag16 ax = loadA_lds(xin, mRow0, k0, c, kh);
                        aR.v = wmma_bf16(ax.v, loadB_w(Wi, n0,          k0, c, kh).v, aR.v);
                        aZ.v = wmma_bf16(ax.v, loadB_w(Wi, Dm + n0,     k0, c, kh).v, aZ.v);
                        aN.v = wmma_bf16(ax.v, loadB_w(Wi, 2 * Dm + n0, k0, c, kh).v, aN.v);
                    }
                }
                const int col = n0 + c;
                const float br  = b_ih[cell * Gm + col]          + b_hh[cell * Gm + col];
                const float bz  = b_ih[cell * Gm + Dm + col]     + b_hh[cell * Gm + Dm + col];
                const float bin = b_ih[cell * Gm + 2 * Dm + col];
                const float bhn = b_hh[cell * Gm + 2 * Dm + col];
                for (int v = 0; v < 8; ++v) {
                    const int r = mRow0 + v + 8 * kh;
                    float* srow = st + (size_t)(rowBase + r) * Dm;
                    float rg = 1.f / (1.f + __expf(-(aR.f[v] + br)));
                    float zg = 1.f / (1.f + __expf(-(aZ.f[v] + bz)));
                    float ng = tanhf(aN.f[v] + bin + rg * (aH.f[v] + bhn));
                    float hn = (1.f - zg) * ng + zg * srow[col];
                    srow[col] = hn;                         // f32 master state (L2-resident)
                    xout[r * STR + col] = f2bf(hn);         // bf16 activation for next layer
                }
            }
        }
        __syncthreads();
        // lin layer: hc = tanh(h3 @ lin_w.T + b)   (h3 is in bufA after cell 2) -> bufB (bf16)
#pragma unroll 1
        for (int j = 0; j < 4; ++j) {
            const int n0 = (jpart * 4 + j) * 16;
            Acc8 a;
            for (int v = 0; v < 8; ++v) a.f[v] = 0.f;
#pragma unroll 2
            for (int kk = 0; kk < 8; ++kk) {
                Frag16 ax = loadA_lds(bufA, mRow0, kk * 32, c, kh);
                a.v = wmma_bf16(ax.v, loadB_w(wlin, n0, kk * 32, c, kh).v, a.v);
            }
            const int col = n0 + c;
            const float bb = lin_b[col];
            for (int v = 0; v < 8; ++v) {
                const int r = mRow0 + v + 8 * kh;
                bufB[r * STR + col] = f2bf(tanhf(a.f[v] + bb));
            }
        }
        __syncthreads();
        // heads: one thread per (row, k); N=10 is below WMMA shape -> f32 VALU
        if (tid < 2 * BT) {
            const int row = tid >> 1;
            const int k   = tid & 1;
            const unsigned short* hc = bufB + row * STR;
            float z1[10];
#pragma unroll 1
            for (int h = 0; h < 10; ++h) {
                const float* wr = hw1 + (size_t)(k * 10 + h) * Dm;
                float s = hb1[k * 10 + h];
#pragma unroll 4
                for (int d = 0; d < Dm; d += 2) {
                    unsigned pp = *(const unsigned*)(hc + d);          // 2 bf16
                    s += __uint_as_float(pp << 16) * wr[d];
                    s += __uint_as_float(pp & 0xffff0000u) * wr[d + 1];
                }
                z1[h] = tanhf(s);
            }
            float* op = out + (size_t)k * Bsz * Tn * 3 + ((size_t)(rowBase + row) * Tn + t) * 3;
#pragma unroll
            for (int o = 0; o < 3; ++o) {
                float s = hb2[k * 3 + o];
#pragma unroll
                for (int h = 0; h < 10; ++h) s += z1[h] * hw2[(k * 3 + o) * 10 + h];
                op[o] = s;
            }
        }
        // loop-top __syncthreads() of next timestep protects buffer reuse
    }
}

// ---------------- host launcher ----------------
extern "C" void kernel_launch(void* const* d_in, const int* in_sizes, int n_in,
                              void* d_out, int out_size, void* d_ws, size_t ws_size,
                              hipStream_t stream) {
    const float* X     = (const float*)d_in[0];
    const float* h0    = (const float*)d_in[1];
    const float* w_ih0 = (const float*)d_in[2];
    const float* w_ihR = (const float*)d_in[3];
    const float* w_hh  = (const float*)d_in[4];
    const float* b_ih  = (const float*)d_in[5];
    const float* b_hh  = (const float*)d_in[6];
    const float* lin_w = (const float*)d_in[7];
    const float* lin_b = (const float*)d_in[8];
    const float* hw1   = (const float*)d_in[9];
    const float* hb1   = (const float*)d_in[10];
    const float* hw2   = (const float*)d_in[11];
    const float* hb2   = (const float*)d_in[12];
    float* out = (float*)d_out;

    // workspace layout (all 16B aligned): f32 state | f32 gi0 | bf16 whh | bf16 wih | bf16 lin_w
    char* ws = (char*)d_ws;
    float* state = (float*)ws;            ws += (size_t)3 * Bsz * Dm * sizeof(float);   // 6 MB
    float* gi0   = (float*)ws;            ws += (size_t)Bsz * Gm * sizeof(float);       // 6 MB
    unsigned short* whh  = (unsigned short*)ws; ws += (size_t)3 * Gm * Dm * 2;          // 1.1 MB
    unsigned short* wih  = (unsigned short*)ws; ws += (size_t)2 * Gm * Dm * 2;          // 0.8 MB
    unsigned short* wlin = (unsigned short*)ws;                                         // 0.13 MB

    prep_weights<<<1048576 / 256, 256, 0, stream>>>(w_hh, w_ihR, lin_w, whh, wih, wlin);
    init_state<<<(3 * Bsz * Dm) / 256, 256, 0, stream>>>(h0, state);
    gi0_kernel<<<(Bsz * Gm) / 256, 256, 0, stream>>>(X, w_ih0, gi0);

    const size_t smem = (size_t)3 * BT * STR * sizeof(unsigned short);   // 50,688 B
    traj_main<<<Bsz / BT, 256, smem, stream>>>(gi0, b_ih, b_hh, lin_b,
                                               hw1, hb1, hw2, hb2,
                                               whh, wih, wlin, state, out);
}
